// Attention_36696200577693
// MI455X (gfx1250) — matrix operand order repaired and checked by reading, not compiled, not run on hardware
//
#include <hip/hip_runtime.h>
#include <hip/hip_bf16.h>
#include <stdint.h>

// ---------------------------------------------------------------------------
// Problem constants: N=8, L=1024, D=256, H=8, R=32, DROPOUT=0.1
// out[n,l,d] = data[n,l,d] * (1 + sum_h w_out[h] * softmax_diag[n,h,l])
// ---------------------------------------------------------------------------

typedef __bf16 bf16;
typedef bf16  v16bf __attribute__((ext_vector_type(16)));
typedef bf16  v8bf  __attribute__((ext_vector_type(8)));
typedef float v8f   __attribute__((ext_vector_type(8)));
typedef unsigned int v4u __attribute__((ext_vector_type(4)));
typedef int   v8i   __attribute__((ext_vector_type(8)));
typedef int   v4i   __attribute__((ext_vector_type(4)));

union BF16x16 { v16bf v; v8bf h[2]; };

// float -> bf16 round-to-nearest-even (stored as raw ushort)
__device__ __forceinline__ unsigned short f2bf(float f) {
  unsigned int u = __float_as_uint(f);
  return (unsigned short)((u + 0x7FFFu + ((u >> 16) & 1u)) >> 16);
}

// Stateless hash RNG emulating the shared [L,L] dropout keep mask
// (keep with prob 1 - 0.1). Deterministic per (l,x).
__device__ __forceinline__ bool rnd_keep(int l, int x) {
  unsigned int s = (unsigned int)(l * 1024 + x) * 0x9E3779B9u;
  s ^= s >> 16; s *= 0x7feb352du;
  s ^= s >> 15; s *= 0x846ca68bu;
  s ^= s >> 16;
  return s > 429496729u;   // u01 > DROPOUT
}

// ---------------------------------------------------------------------------
// Prep kernels
// ---------------------------------------------------------------------------
__global__ void __launch_bounds__(256)
k_data_bf16(const float* __restrict__ in, unsigned short* __restrict__ out) {
  int t = blockIdx.x * 256 + threadIdx.x;           // N*L*D/4 = 524288 threads
  float4 v = ((const float4*)in)[t];
  ushort4 o;
  o.x = f2bf(v.x); o.y = f2bf(v.y); o.z = f2bf(v.z); o.w = f2bf(v.w);
  ((ushort4*)out)[t] = o;
}

// weights [H,D,R] f32 -> WT [H,R,D] bf16 so WMMA B-fragments are contiguous
__global__ void __launch_bounds__(256)
k_weights_T(const float* __restrict__ W, unsigned short* __restrict__ WT) {
  int t = blockIdx.x * 256 + threadIdx.x;           // H*D*R = 65536
  int h = t >> 13, rem = t & 8191;
  int d = rem >> 5, r = rem & 31;
  WT[((h * 32 + r) << 8) + d] = f2bf(W[t]);
}

// ---------------------------------------------------------------------------
// Kernel B: proj[pair][l][r] = data_n (L x 256) @ W_h (256 x 32), bf16 WMMA.
// One 16x16 output tile per wave; 8 k-steps of v_wmma_f32_16x16x32_bf16.
// ---------------------------------------------------------------------------
__global__ void __launch_bounds__(256)
k_proj(const unsigned short* __restrict__ dataBF,
       const unsigned short* __restrict__ wtBF,
       unsigned short* __restrict__ projBF) {
  const int wave = threadIdx.x >> 5, lane = threadIdx.x & 31;
  const int p = lane & 15, hi = lane >> 4;
  const int t = blockIdx.x * 8 + wave;              // 8192 tiles
  const int pair = t >> 7;                          // n*8+h
  const int rem = t & 127;
  const int lt = rem >> 1, ct = rem & 1;
  const int n = pair >> 3, h = pair & 7;

  const bf16* A = (const bf16*)dataBF + ((size_t)n * 1024 + lt * 16 + p) * 256;
  const bf16* B = (const bf16*)wtBF  + ((size_t)h * 32  + ct * 16 + p) * 256;

  v8f acc = {};
#pragma unroll
  for (int k = 0; k < 256; k += 32) {
    BF16x16 af, bfm;
    // A 16x32 bf16 layout: lane p row M=p; lo half K {0..7,16..23}, hi {8..15,24..31}
    af.h[0]  = *(const v8bf*)(A + k + hi * 8);
    af.h[1]  = *(const v8bf*)(A + k + 16 + hi * 8);
    // B 32x16 bf16 layout: lane p col N=p; lo half K {0..7,16..23}, hi {8..15,24..31}
    bfm.h[0] = *(const v8bf*)(B + k + hi * 8);
    bfm.h[1] = *(const v8bf*)(B + k + 16 + hi * 8);
    acc = __builtin_amdgcn_wmma_f32_16x16x32_bf16(false, af.v, false, bfm.v,
                                                  (short)0, acc, false, false);
  }
  // C layout: slot v -> (M = v + 8*hi, N = p)
  unsigned short* out = projBF + (size_t)pair * 32768 + (lt * 16) * 32 + ct * 16 + p;
#pragma unroll
  for (int v = 0; v < 8; ++v) out[(v + 8 * hi) * 32] = f2bf(acc[v]);
}

// ---------------------------------------------------------------------------
// Kernel C: fused S = P*P^T, dropout mask, softmax row stats, diagonal only.
// One (n,h) pair per block; whole P (64KB) TDM-loaded into LDS; each of the
// 8 waves owns 16 rows, loops 64 x-tiles; two passes (max, then sum-exp).
// ---------------------------------------------------------------------------
__global__ void __launch_bounds__(256)
k_score_softmax(const unsigned short* __restrict__ projBF,
                const unsigned char* __restrict__ maskp,
                const float* __restrict__ wout,
                float* __restrict__ gpart) {
  __shared__ unsigned short Pbuf[32768];            // 64 KB: P[1024][32] bf16
  const int pair = blockIdx.y;
  const int n = pair >> 3, h = pair & 7;
  const int wave = threadIdx.x >> 5, lane = threadIdx.x & 31;
  const int p = lane & 15, hi = lane >> 4;

  // --- TDM bulk copy: 64KB global -> LDS (1D: 8192 x 8B), wave 0 issues ---
  if (wave == 0) {
    unsigned long long ga =
        (unsigned long long)(uintptr_t)(projBF + (size_t)pair * 32768);
    unsigned int lds = (unsigned int)(uintptr_t)(&Pbuf[0]);
    v4u g0;
    g0[0] = 1u;                                     // count=1, user descriptor
    g0[1] = lds;                                    // lds_addr
    g0[2] = (unsigned int)ga;                       // global_addr[31:0]
    g0[3] = ((unsigned int)(ga >> 32) & 0x01FFFFFFu) | 0x80000000u; // addr hi | type=2
    v8i g1;
    g1[0] = 0x00030000;  // data_size=3 (8B), no multicast/pad/iterate
    g1[1] = 0x20000000;  // tensor_dim0 = 8192 (low16 in bits 63:48)
    g1[2] = 0x00010000;  // tensor_dim0 hi=0, tensor_dim1 = 1
    g1[3] = 0x20000000;  // tile_dim0 = 8192
    g1[4] = 1;           // tile_dim1 = 1, tile_dim2 = 0
    g1[5] = 8192;        // tensor_dim0_stride lo
    g1[6] = 0x20000000;  // stride0 hi=0, tensor_dim1_stride lo16 = 8192
    g1[7] = 0;
    v4i zz4 = {0, 0, 0, 0};
    v8i zz8 = {0, 0, 0, 0, 0, 0, 0, 0};
    // 6-arg toolchain variant: (g0, g1, g2, g3, extra, cpol)
    __builtin_amdgcn_tensor_load_to_lds(g0, g1, zz4, zz4, zz8, 0);
    __builtin_amdgcn_s_wait_tensorcnt(0);
  }
  __syncthreads();

  const bf16* Pl = (const bf16*)Pbuf;
  const int l0 = blockIdx.x * 128 + wave * 16;      // this wave's 16 rows

  // A fragment (16 rows x K=32) held for the whole kernel
  BF16x16 af;
  {
    const bf16* ar = Pl + (l0 + p) * 32;
    af.h[0] = *(const v8bf*)(ar + hi * 8);
    af.h[1] = *(const v8bf*)(ar + 16 + hi * 8);
  }
  const unsigned char* mrow = maskp + (size_t)n * 1024;

  float m[8];
#pragma unroll
  for (int v = 0; v < 8; ++v) m[v] = -INFINITY;

  // ---- pass 1: per-row max over kept entries ----
  for (int xt = 0; xt < 64; ++xt) {
    BF16x16 bfm;
    const bf16* br = Pl + (xt * 16 + p) * 32;       // B column N=p is P row x
    bfm.h[0] = *(const v8bf*)(br + hi * 8);
    bfm.h[1] = *(const v8bf*)(br + 16 + hi * 8);
    v8f c = {};
    c = __builtin_amdgcn_wmma_f32_16x16x32_bf16(false, af.v, false, bfm.v,
                                                (short)0, c, false, false);
    const int x = xt * 16 + p;
    const bool mk = mrow[x] != 0;
#pragma unroll
    for (int v = 0; v < 8; ++v) {
      const int l = l0 + v + 8 * hi;
      const bool kp = (l == x) || (mk && rnd_keep(l, x));
      if (kp) m[v] = fmaxf(m[v], c[v]);
    }
  }
#pragma unroll
  for (int v = 0; v < 8; ++v)
#pragma unroll
    for (int o = 1; o < 16; o <<= 1)
      m[v] = fmaxf(m[v], __shfl_xor(m[v], o, 32));

  // ---- pass 2: sum of exp + diagonal capture (recompute via WMMA) ----
  float s[8], dg[8];
#pragma unroll
  for (int v = 0; v < 8; ++v) { s[v] = 0.f; dg[v] = 0.f; }
  for (int xt = 0; xt < 64; ++xt) {
    BF16x16 bfm;
    const bf16* br = Pl + (xt * 16 + p) * 32;
    bfm.h[0] = *(const v8bf*)(br + hi * 8);
    bfm.h[1] = *(const v8bf*)(br + 16 + hi * 8);
    v8f c = {};
    c = __builtin_amdgcn_wmma_f32_16x16x32_bf16(false, af.v, false, bfm.v,
                                                (short)0, c, false, false);
    const int x = xt * 16 + p;
    const bool mk = mrow[x] != 0;
#pragma unroll
    for (int v = 0; v < 8; ++v) {
      const int l = l0 + v + 8 * hi;
      const bool kp = (l == x) || (mk && rnd_keep(l, x));
      const float e = kp ? c[v] : -INFINITY;        // exp(-inf)=0
      s[v] += __expf(e - m[v]);
      if (l == x) dg[v] = c[v];                     // diagonal always kept
    }
  }
#pragma unroll
  for (int v = 0; v < 8; ++v)
#pragma unroll
    for (int o = 1; o < 16; o <<= 1) {
      s[v]  += __shfl_xor(s[v], o, 32);
      dg[v] += __shfl_xor(dg[v], o, 32);
    }

  if (p == 0) {                                     // lane 0 and lane 16
    const float wh = wout[h];
#pragma unroll
    for (int v = 0; v < 8; ++v) {
      const int l = l0 + v + 8 * hi;
      gpart[(size_t)pair * 1024 + l] = wh * __expf(dg[v] - m[v]) / s[v];
    }
  }
}

// ---------------------------------------------------------------------------
// Kernel D: out = data * (1 + sum_h gpart[n,h,l])
// ---------------------------------------------------------------------------
__global__ void __launch_bounds__(256)
k_final(const float* __restrict__ data, const float* __restrict__ gpart,
        float* __restrict__ out) {
  int t = blockIdx.x * 256 + threadIdx.x;           // N*L*D/4 threads
  int nl = t >> 6;                                  // D/4 = 64
  int nn = nl >> 10, l = nl & 1023;
  float sc = 1.0f;
#pragma unroll
  for (int h = 0; h < 8; ++h) sc += gpart[((size_t)(nn * 8 + h)) * 1024 + l];
  float4 v = ((const float4*)data)[t];
  v.x *= sc; v.y *= sc; v.z *= sc; v.w *= sc;
  ((float4*)out)[t] = v;
}

// ---------------------------------------------------------------------------
extern "C" void kernel_launch(void* const* d_in, const int* in_sizes, int n_in,
                              void* d_out, int out_size, void* d_ws, size_t ws_size,
                              hipStream_t stream) {
  const float* data = (const float*)d_in[0];          // [8,1024,256] f32
  const float* W    = (const float*)d_in[1];          // [8,256,32]  f32
  const float* wout = (const float*)d_in[2];          // [8]         f32
  const unsigned char* mask = (const unsigned char*)d_in[3]; // [8,1024] bool
  float* out = (float*)d_out;                          // [8,1024,256] f32

  char* ws = (char*)d_ws;
  unsigned short* dataBF = (unsigned short*)(ws);                    // 4 MB
  unsigned short* wtBF   = (unsigned short*)(ws + 0x400000);         // 128 KB
  unsigned short* projBF = (unsigned short*)(ws + 0x420000);         // 4 MB
  float*          gpart  = (float*)        (ws + 0x820000);          // 256 KB

  k_data_bf16<<<2048, 256, 0, stream>>>(data, dataBF);
  k_weights_T<<<256, 256, 0, stream>>>(W, wtBF);
  k_proj<<<1024, 256, 0, stream>>>(dataBF, wtBF, projBF);
  dim3 gC(8, 64);
  k_score_softmax<<<gC, 256, 0, stream>>>(projBF, mask, wout, gpart);
  k_final<<<2048, 256, 0, stream>>>(data, gpart, out);
}
